// OpenAIPrivacyFilterAttention_34471407518011
// MI455X (gfx1250) — matrix operand order repaired and checked by reading, not compiled, not run on hardware
//
#include <hip/hip_runtime.h>
#include <hip/hip_bf16.h>

typedef __attribute__((ext_vector_type(16))) _Float16 v16h;
typedef __attribute__((ext_vector_type(8)))  _Float16 v8h;
typedef __attribute__((ext_vector_type(8)))  float    v8f;
typedef unsigned int u32x4 __attribute__((ext_vector_type(4)));
typedef int          i32x4 __attribute__((ext_vector_type(4)));
typedef int          i32x8 __attribute__((ext_vector_type(8)));

#define SEQ 2048
#define MODEL 2048
#define NH 32
#define NKV 8
#define HD 64
#define NQKV 3072
#define SCALE_C 0.35355339059327379f   // 64^-0.25
#define NEG_C  -1.0e9f

#if defined(__has_builtin)
#if __has_builtin(__builtin_amdgcn_tensor_load_to_lds) && __has_builtin(__builtin_amdgcn_s_wait_tensorcnt)
#define HAVE_TDM 1
#endif
#endif

__device__ __forceinline__ v16h cat8(v8h lo, v8h hi) {
  return __builtin_shufflevector(lo, hi, 0,1,2,3,4,5,6,7,8,9,10,11,12,13,14,15);
}

__device__ __forceinline__ v8f wmma16(v16h a, v16h b, v8f c) {
  return __builtin_amdgcn_wmma_f32_16x16x32_f16(false, a, false, b, (short)0, c, false, false);
}

#if defined(HAVE_TDM)
// Issue one TDM load of a 128-row x 64-byte tile (row stride = rowStrideUnits*8 bytes)
// from global memory into LDS at lds_off. data_size = 8B units; 2D descriptor,
// groups 2/3 zeroed (2D tile -> unused).
__device__ __forceinline__ void tdm_load_tile(unsigned lds_off, const void* gptr,
                                              unsigned rowStrideUnits) {
  unsigned long long ga = (unsigned long long)(size_t)gptr;
  u32x4 g0;
  g0[0] = 1u;                                              // count=1, user descriptor
  g0[1] = lds_off;                                         // LDS byte address
  g0[2] = (unsigned)(ga & 0xFFFFFFFFu);                    // global_addr[31:0]
  g0[3] = (unsigned)((ga >> 32) & 0x01FFFFFFu) | (2u << 30); // global_addr[56:32] | type=2
  unsigned td0 = rowStrideUnits;                           // tensor_dim0 (8B units)
  unsigned td1 = 128u;                                     // tensor_dim1 (rows)
  i32x8 g1;
  g1[0] = (int)(3u << 16);                                 // workgroup_mask=0, data_size=3 (8B)
  g1[1] = (int)((td0 & 0xFFFFu) << 16);                    // abar=0 | tensor_dim0[15:0]
  g1[2] = (int)(((td0 >> 16) & 0xFFFFu) | ((td1 & 0xFFFFu) << 16));
  g1[3] = (int)(((td1 >> 16) & 0xFFFFu) | (8u << 16));     // tensor_dim1 hi | tile_dim0=8 (64B)
  g1[4] = (int)128u;                                       // tile_dim1=128 | tile_dim2=0
  g1[5] = (int)rowStrideUnits;                             // tensor_dim0_stride[31:0]
  g1[6] = 0;                                               // stride hi | dim1_stride lo
  g1[7] = 0;
  i32x4 z4 = {0, 0, 0, 0};
#if __clang_major__ >= 23
  i32x8 z8 = {0, 0, 0, 0, 0, 0, 0, 0};
  __builtin_amdgcn_tensor_load_to_lds(g0, g1, z4, z4, z8, 0);
#else
  __builtin_amdgcn_tensor_load_to_lds(g0, g1, z4, z4, 0);
#endif
}
#endif

// ---------------- conversion / packing kernels ----------------

__global__ void k_f32_to_f16(const float* __restrict__ x, _Float16* __restrict__ y, int n) {
  int i = blockIdx.x * 256 + threadIdx.x;
  if (i < n) y[i] = (_Float16)x[i];
}

// wT[n][m] (K-major) from Wq[m][2048] | Wk[m][512] | Wv[m][512]
__global__ void k_pack_wqkv(const float* __restrict__ Wq, const float* __restrict__ Wk,
                            const float* __restrict__ Wv, _Float16* __restrict__ wT) {
  long id = (long)blockIdx.x * 256 + threadIdx.x;      // 3072*2048 threads
  int n = (int)(id >> 11);
  int m = (int)(id & 2047);
  float v;
  if (n < 2048)      v = Wq[(long)m * 2048 + n];
  else if (n < 2560) v = Wk[(long)m * 512 + (n - 2048)];
  else               v = Wv[(long)m * 512 + (n - 2560)];
  wT[id] = (_Float16)v;
}

// woT[n=m_out][k] from Wo[k][m_out]  (Wo flat: [H*D=2048][M=2048])
__global__ void k_pack_wo(const float* __restrict__ Wo, _Float16* __restrict__ wT) {
  long id = (long)blockIdx.x * 256 + threadIdx.x;      // 2048*2048
  int n = (int)(id >> 11);
  int k = (int)(id & 2047);
  wT[id] = (_Float16)Wo[(long)k * 2048 + n];
}

__global__ void k_pack_bias(const float* __restrict__ bq, const float* __restrict__ bk,
                            const float* __restrict__ bv, float* __restrict__ b) {
  int i = blockIdx.x * 256 + threadIdx.x;
  if (i >= NQKV) return;
  if (i < 2048)      b[i] = bq[i];
  else if (i < 2560) b[i] = bk[i - 2048];
  else               b[i] = bv[i - 2560];
}

// ---------------- WMMA GEMM: C[S][N] = A[S][K](f16) * Bt[N][K](f16) + bias[N] ----------------
// macro tile 128x128, K-step 32, 256 threads = 8 waves, each wave 2x4 16x16 tiles.
// TDM path: double-buffered LDS; stage i+1's DMA overlaps stage i's WMMAs.
// Tensor ops complete in-order per wave, so s_wait_tensorcnt<=2 means stage i is resident.

__global__ __launch_bounds__(256) void k_gemm(const _Float16* __restrict__ A,
                                              const _Float16* __restrict__ Bt,
                                              const float* __restrict__ bias,
                                              float* __restrict__ C, int N, int K) {
  __shared__ __align__(16) _Float16 As[2 * 128 * 32];
  __shared__ __align__(16) _Float16 Bs[2 * 128 * 32];
  const int t = threadIdx.x;
  const int rowBase = blockIdx.y * 128;
  const int colBase = blockIdx.x * 128;
  const int wave = t >> 5, lane = t & 31;
  const int rw = (wave & 3) * 32;
  const int cw = (wave >> 2) * 64;
  const int lrow = lane & 15, lhi = lane >> 4;

  v8f acc[2][4];
#pragma unroll
  for (int rt = 0; rt < 2; ++rt)
#pragma unroll
    for (int ct = 0; ct < 4; ++ct)
#pragma unroll
      for (int e = 0; e < 8; ++e) acc[rt][ct][e] = 0.0f;

#if defined(HAVE_TDM)
  const unsigned asOff = (unsigned)(size_t)(const void*)&As[0];
  const unsigned bsOff = (unsigned)(size_t)(const void*)&Bs[0];
  const unsigned strideUnits = (unsigned)(K >> 2);   // K halves * 2B / 8B
  const int nIter = K >> 5;

  if (t < 32) {                                      // wave 0 drives the TDM
    tdm_load_tile(asOff, &A[(long)rowBase * K], strideUnits);
    tdm_load_tile(bsOff, &Bt[(long)colBase * K], strideUnits);
  }
  for (int it = 0; it < nIter; ++it) {
    const int cur = it & 1;
    if (t < 32) {
      if (it + 1 < nIter) {                          // prefetch next stage into other buffer
        const int k0n = (it + 1) << 5;
        const unsigned nb = (unsigned)((it + 1) & 1) * (128u * 32u * 2u);
        tdm_load_tile(asOff + nb, &A[(long)rowBase * K + k0n], strideUnits);
        tdm_load_tile(bsOff + nb, &Bt[(long)colBase * K + k0n], strideUnits);
        __builtin_amdgcn_s_wait_tensorcnt(2);        // stage 'it' (2 older ops) complete
      } else {
        __builtin_amdgcn_s_wait_tensorcnt(0);
      }
    }
    __syncthreads();

    const _Float16* Asb = &As[cur * (128 * 32)];
    const _Float16* Bsb = &Bs[cur * (128 * 32)];
    v16h a[2], b[4];
#pragma unroll
    for (int rt = 0; rt < 2; ++rt) {
      const _Float16* p = &Asb[(rw + rt * 16 + lrow) * 32 + lhi * 8];
      a[rt] = cat8(*(const v8h*)p, *(const v8h*)(p + 16));
    }
#pragma unroll
    for (int ct = 0; ct < 4; ++ct) {
      const _Float16* p = &Bsb[(cw + ct * 16 + lrow) * 32 + lhi * 8];
      b[ct] = cat8(*(const v8h*)p, *(const v8h*)(p + 16));
    }
#pragma unroll
    for (int rt = 0; rt < 2; ++rt)
#pragma unroll
      for (int ct = 0; ct < 4; ++ct)
        acc[rt][ct] = wmma16(a[rt], b[ct], acc[rt][ct]);
    __syncthreads();                                 // buffer free before stage it+2 overwrites
  }
#else
  for (int k0 = 0; k0 < K; k0 += 32) {
#pragma unroll
    for (int i = 0; i < 2; ++i) {
      int idx = t + i * 256;
      int r = idx >> 2, seg = (idx & 3) * 8;
      *(v8h*)&As[r * 32 + seg] = *(const v8h*)&A[(long)(rowBase + r) * K + k0 + seg];
      *(v8h*)&Bs[r * 32 + seg] = *(const v8h*)&Bt[(long)(colBase + r) * K + k0 + seg];
    }
    __syncthreads();
    if (k0 + 32 < K) {   // emits global_prefetch_b8 for the next K-slice
      __builtin_prefetch(&A[(long)(rowBase + (t >> 1)) * K + k0 + 32], 0, 1);
      __builtin_prefetch(&Bt[(long)(colBase + (t >> 1)) * K + k0 + 32], 0, 1);
    }
    v16h a[2], b[4];
#pragma unroll
    for (int rt = 0; rt < 2; ++rt) {
      const _Float16* p = &As[(rw + rt * 16 + lrow) * 32 + lhi * 8];
      a[rt] = cat8(*(const v8h*)p, *(const v8h*)(p + 16));
    }
#pragma unroll
    for (int ct = 0; ct < 4; ++ct) {
      const _Float16* p = &Bs[(cw + ct * 16 + lrow) * 32 + lhi * 8];
      b[ct] = cat8(*(const v8h*)p, *(const v8h*)(p + 16));
    }
#pragma unroll
    for (int rt = 0; rt < 2; ++rt)
#pragma unroll
      for (int ct = 0; ct < 4; ++ct)
        acc[rt][ct] = wmma16(a[rt], b[ct], acc[rt][ct]);
    __syncthreads();
  }
#endif

#pragma unroll
  for (int rt = 0; rt < 2; ++rt)
#pragma unroll
    for (int ct = 0; ct < 4; ++ct) {
      int n = colBase + cw + ct * 16 + lrow;
      float bn = bias[n];
#pragma unroll
      for (int r = 0; r < 8; ++r) {
        int m = rowBase + rw + rt * 16 + r + 8 * lhi;
        C[(long)m * N + n] = acc[rt][ct][r] + bn;
      }
    }
}

// ---------------- RoPE + scale + f16 conversion ----------------

__global__ void k_rope_q(const float* __restrict__ qkv, _Float16* __restrict__ q16) {
  int id = blockIdx.x * 256 + threadIdx.x;   // S*H*32
  int i = id & 31, h = (id >> 5) & 31, s = id >> 10;
  const float* base = qkv + (long)s * NQKV + h * 64;
  float e = base[2 * i], o = base[2 * i + 1];
  float invf = __powf(150000.0f, -((float)(2 * i) / 64.0f));
  float sn, cs;
  __sincosf((float)s * invf, &sn, &cs);
  _Float16* out = q16 + ((long)s * 32 + h) * 64;
  out[2 * i]     = (_Float16)((e * cs - o * sn) * SCALE_C);
  out[2 * i + 1] = (_Float16)((o * cs + e * sn) * SCALE_C);
}

__global__ void k_rope_k(const float* __restrict__ qkv, _Float16* __restrict__ k16) {
  int id = blockIdx.x * 256 + threadIdx.x;   // S*HKV*32
  int i = id & 31, hk = (id >> 5) & 7, s = id >> 8;
  const float* base = qkv + (long)s * NQKV + 2048 + hk * 64;
  float e = base[2 * i], o = base[2 * i + 1];
  float invf = __powf(150000.0f, -((float)(2 * i) / 64.0f));
  float sn, cs;
  __sincosf((float)s * invf, &sn, &cs);
  _Float16* out = k16 + ((long)s * 8 + hk) * 64;
  out[2 * i]     = (_Float16)((e * cs - o * sn) * SCALE_C);
  out[2 * i + 1] = (_Float16)((o * cs + e * sn) * SCALE_C);
}

// vT[hk][d][s]  (transposed so PV B-operand K-runs are contiguous)
__global__ void k_conv_v(const float* __restrict__ qkv, _Float16* __restrict__ vT) {
  int id = blockIdx.x * 256 + threadIdx.x;   // S*HKV*64
  int d = id & 63, hk = (id >> 6) & 7, s = id >> 9;
  vT[((long)(hk * 64 + d) << 11) + s] = (_Float16)qkv[(long)s * NQKV + 2560 + hk * 64 + d];
}

// ---------------- sliding-window attention with sinks (1 wave / (qtile, head)) ----------------

__global__ __launch_bounds__(32) void k_attn(const _Float16* __restrict__ q16,
                                             const _Float16* __restrict__ k16,
                                             const _Float16* __restrict__ vT,
                                             const float* __restrict__ sinks,
                                             _Float16* __restrict__ o16) {
  __shared__ __align__(16) float    Sl[16 * 288];
  __shared__ __align__(16) _Float16 Pl[16 * 288];
  __shared__ float invs[16];

  const int lane = threadIdx.x;
  const int qt = blockIdx.x, h = blockIdx.y, hkv = h >> 2;
  const int qBase = qt * 16;
  const int lrow = lane & 15, lhi = lane >> 4;
  const int ktLo = (qt >= 8) ? qt - 8 : 0;
  const int ktHi = (qt + 8 <= 127) ? qt + 8 : 127;
  const int nkt = ktHi - ktLo + 1;
  const int nkeys = nkt * 16;
  const int nk32 = (nkt + 1) >> 1;
  const int nkPad = nk32 * 32;
  const int kh = lhi * 8;

  // Q as A-operand (reused across all key tiles): lane=row, two 16B runs per K=32 step
  const _Float16* qp = q16 + ((long)(qBase + lrow) * 32 + h) * 64;
  v16h a0 = cat8(*(const v8h*)(qp + kh),      *(const v8h*)(qp + 16 + kh));
  v16h a1 = cat8(*(const v8h*)(qp + 32 + kh), *(const v8h*)(qp + 48 + kh));

  // ---- scores: S = q k^T, masked, into LDS ----
  for (int kt = ktLo; kt <= ktHi; ++kt) {
    int key = kt * 16 + lrow;
    const _Float16* kp = k16 + ((long)key * 8 + hkv) * 64 + lhi * 8;
    v16h b0 = cat8(*(const v8h*)kp,        *(const v8h*)(kp + 16));
    v16h b1 = cat8(*(const v8h*)(kp + 32), *(const v8h*)(kp + 48));
    v8f c;
#pragma unroll
    for (int e = 0; e < 8; ++e) c[e] = 0.0f;
    c = wmma16(a0, b0, c);
    c = wmma16(a1, b1, c);
    int col = (kt - ktLo) * 16 + lrow;
#pragma unroll
    for (int r = 0; r < 8; ++r) {
      int m = r + 8 * lhi;
      int dd = (qBase + m) - key;
      if (dd < 0) dd = -dd;
      Sl[m * 288 + col] = (dd <= 128) ? c[r] : NEG_C;
    }
  }
  __syncthreads();

  // ---- softmax with sink (2 lanes per row; stats replicated via shfl_xor 16) ----
  {
    float sink = sinks[h];
    float mloc = NEG_C;
    for (int kk = lhi; kk < nkeys; kk += 2) mloc = fmaxf(mloc, Sl[lrow * 288 + kk]);
    float mrow = fmaxf(fmaxf(mloc, __shfl_xor(mloc, 16, 32)), sink);
    float sloc = 0.0f;
    for (int kk = lhi; kk < nkeys; kk += 2) {
      float p = __expf(Sl[lrow * 288 + kk] - mrow);
      Pl[lrow * 288 + kk] = (_Float16)p;     // unnormalized; normalize at the end
      sloc += p;
    }
    float sum = sloc + __shfl_xor(sloc, 16, 32) + __expf(sink - mrow);
    if (lhi == 0) invs[lrow] = 1.0f / sum;
    // zero-pad P columns up to the next multiple of 32 keys
    for (int kk = nkeys + lane; kk < nkPad; kk += 32)
      for (int rr = 0; rr < 16; ++rr) Pl[rr * 288 + kk] = (_Float16)0.0f;
  }
  __syncthreads();

  // ---- O = P V  (P from LDS in A layout; V from transposed global, B layout) ----
  v8f o[4];
#pragma unroll
  for (int ct = 0; ct < 4; ++ct)
#pragma unroll
    for (int e = 0; e < 8; ++e) o[ct][e] = 0.0f;

  for (int j = 0; j < nk32; ++j) {
    const _Float16* pp = &Pl[lrow * 288 + j * 32 + kh];
    v16h pa = cat8(*(const v8h*)pp, *(const v8h*)(pp + 16));
#pragma unroll
    for (int ct = 0; ct < 4; ++ct) {
      int d = ct * 16 + lrow;
      const _Float16* vp = vT + ((long)(hkv * 64 + d) << 11) + ktLo * 16 + j * 32 + lhi * 8;
      v16h vb = cat8(*(const v8h*)vp, *(const v8h*)(vp + 16));
      o[ct] = wmma16(pa, vb, o[ct]);
    }
  }

#pragma unroll
  for (int ct = 0; ct < 4; ++ct)
#pragma unroll
    for (int r = 0; r < 8; ++r) {
      int m = r + 8 * lhi;
      o16[((long)(qBase + m) * 32 + h) * 64 + ct * 16 + lrow] = (_Float16)(o[ct][r] * invs[m]);
    }
}

// ---------------- launch ----------------

extern "C" void kernel_launch(void* const* d_in, const int* in_sizes, int n_in,
                              void* d_out, int out_size, void* d_ws, size_t ws_size,
                              hipStream_t stream) {
  (void)in_sizes; (void)n_in; (void)out_size; (void)ws_size;
  const float* hidden = (const float*)d_in[0];
  const float* Wq = (const float*)d_in[1];
  const float* bq = (const float*)d_in[2];
  const float* Wk = (const float*)d_in[3];
  const float* bk = (const float*)d_in[4];
  const float* Wv = (const float*)d_in[5];
  const float* bv = (const float*)d_in[6];
  const float* Wo = (const float*)d_in[7];
  const float* bo = (const float*)d_in[8];
  const float* sinks = (const float*)d_in[9];

  size_t off = 0;
  auto carve = [&](size_t bytes) {
    void* p = (char*)d_ws + off;
    off += (bytes + 255) & ~(size_t)255;
    return p;
  };
  _Float16* hf16    = (_Float16*)carve((size_t)SEQ * MODEL * 2);
  _Float16* wqkvT   = (_Float16*)carve((size_t)NQKV * MODEL * 2);
  float*    biasqkv = (float*)carve((size_t)NQKV * 4);
  float*    qkvf    = (float*)carve((size_t)SEQ * NQKV * 4);
  _Float16* q16     = (_Float16*)carve((size_t)SEQ * NH * HD * 2);
  _Float16* k16     = (_Float16*)carve((size_t)SEQ * NKV * HD * 2);
  _Float16* vT      = (_Float16*)carve((size_t)SEQ * NKV * HD * 2 + 4096); // slack for padded reads
  _Float16* o16     = (_Float16*)carve((size_t)SEQ * NH * HD * 2);
  _Float16* woT     = (_Float16*)carve((size_t)MODEL * MODEL * 2);

  k_f32_to_f16<<<(SEQ * MODEL) / 256, 256, 0, stream>>>(hidden, hf16, SEQ * MODEL);
  k_pack_wqkv<<<(NQKV * MODEL) / 256, 256, 0, stream>>>(Wq, Wk, Wv, wqkvT);
  k_pack_bias<<<(NQKV + 255) / 256, 256, 0, stream>>>(bq, bk, bv, biasqkv);
  k_pack_wo<<<(MODEL * MODEL) / 256, 256, 0, stream>>>(Wo, woT);

  k_gemm<<<dim3(NQKV / 128, SEQ / 128), 256, 0, stream>>>(hf16, wqkvT, biasqkv, qkvf, NQKV, MODEL);

  k_rope_q<<<(SEQ * NH * 32) / 256, 256, 0, stream>>>(qkvf, q16);
  k_rope_k<<<(SEQ * NKV * 32) / 256, 256, 0, stream>>>(qkvf, k16);
  k_conv_v<<<(SEQ * NKV * HD) / 256, 256, 0, stream>>>(qkvf, vT);

  k_attn<<<dim3(SEQ / 16, NH), 32, 0, stream>>>(q16, k16, vT, sinks, o16);

  k_gemm<<<dim3(MODEL / 128, SEQ / 128), 256, 0, stream>>>(o16, woT, bo, (float*)d_out, MODEL, MODEL);
}
